// MultiheadAttention_8443905704482
// MI455X (gfx1250) — compile-verified, matrix operand order checked by reading, not run on hardware
//
#include <hip/hip_runtime.h>
#include <hip/hip_bf16.h>
#include <hip/hip_fp16.h>

// Problem constants (match the JAX reference)
#define B_  4
#define T_  2048
#define D_  1024
#define H_  16
#define HD_ 64

typedef __attribute__((ext_vector_type(16))) _Float16 v16h;
typedef __attribute__((ext_vector_type(8)))  _Float16 v8h;
typedef __attribute__((ext_vector_type(8)))  float    v8f;
typedef __attribute__((ext_vector_type(4)))  float    v4f;

__device__ __forceinline__ v8h ld8h(const _Float16* p) {
    return *reinterpret_cast<const v8h*>(p);
}
__device__ __forceinline__ v4f ld4f(const float* p) {
    return *reinterpret_cast<const v4f*>(p);
}
__device__ __forceinline__ v16h cat8(v8h lo, v8h hi) {
    v16h r;
#pragma unroll
    for (int i = 0; i < 8; ++i) { r[i] = lo[i]; r[8 + i] = hi[i]; }
    return r;
}
__device__ __forceinline__ v8f wmma16(v16h a, v16h b, v8f c) {
    // D = A(16x32 f16) * B(32x16 f16) + C(16x16 f32)
    return __builtin_amdgcn_wmma_f32_16x16x32_f16(
        /*neg_a=*/false, a, /*neg_b=*/false, b,
        /*c_mod=*/(short)0, c, /*reuse_a=*/false, /*reuse_b=*/false);
}
// Wave-relative LDS byte offset of a generic pointer to a __shared__ object
// (flat LDS address truncates to the LDS offset in its low 32 bits, ISA 10.2).
__device__ __forceinline__ unsigned lds_off(const void* p) {
    return (unsigned)(unsigned long long)p;
}
// gfx1250 async copy global -> LDS, 16B per lane, tracked with ASYNCcnt.
__device__ __forceinline__ void async_g2l_b128(unsigned ldsoff, const void* g) {
    unsigned long long ga = (unsigned long long)g;
    asm volatile("global_load_async_to_lds_b128 %0, %1, off"
                 :: "v"(ldsoff), "v"(ga) : "memory");
}

// ---------------------------------------------------------------------------
// Kernel 1: convert the four DxD weight matrices to f16 (packed consecutively)
// ---------------------------------------------------------------------------
__global__ void __launch_bounds__(256) cvt_w_kernel(
    const float* __restrict__ Wq, const float* __restrict__ Wk,
    const float* __restrict__ Wv, const float* __restrict__ Wo,
    _Float16* __restrict__ out)
{
    size_t i = (size_t)blockIdx.x * blockDim.x + threadIdx.x;  // 4*D*D threads
    size_t m = i >> 20;                 // D*D == 1<<20
    size_t off = i & ((1u << 20) - 1);
    const float* s = (m == 0) ? Wq : ((m == 1) ? Wk : ((m == 2) ? Wv : Wo));
    out[i] = (_Float16)s[off];
}

// ---------------------------------------------------------------------------
// Kernel 2: input projections  y = x @ W^T   (x f32 -> f16 heads)
//  mat 0: q -> qh [b,h,t,hd]   mat 1: k -> kh [b,h,t,hd]   mat 2: v -> vht [b,h,hd,t]
//  Wave tile: 16 rows x 64 cols, K = 1024 in 32 WMMA steps.
// ---------------------------------------------------------------------------
__global__ void __launch_bounds__(128) proj_kernel(
    const float* __restrict__ Xq, const float* __restrict__ Xk,
    const float* __restrict__ Xv, const _Float16* __restrict__ W16base,
    _Float16* __restrict__ qh, _Float16* __restrict__ kh,
    _Float16* __restrict__ vht)
{
    const int lane = threadIdx.x & 31;
    const int warp = threadIdx.x >> 5;
    const int wave = blockIdx.x * 4 + warp;
    const int MT = (B_ * T_) / 16;      // 512 row tiles
    const int m0 = (wave % MT) * 16;
    const int n0 = (wave / MT) * 64;
    const int mat = blockIdx.y;

    const float*    X = (mat == 0) ? Xq : ((mat == 1) ? Xk : Xv);
    const _Float16* W = W16base + (size_t)mat * D_ * D_;

    const int l15  = lane & 15;              // A row / B col within tile
    const int koff = (lane & 16) ? 8 : 0;    // ISA 16-bit A/B K-chunk select
    const int hi8  = (lane & 16) ? 8 : 0;    // C layout: upper 8 rows

    v8f c[4] = {v8f{}, v8f{}, v8f{}, v8f{}};

    const float* xr = X + (size_t)(m0 + l15) * D_;
#pragma unroll 1
    for (int ks = 0; ks < D_ / 32; ++ks) {
        const int k0 = ks * 32 + koff;
        v4f f0 = ld4f(xr + k0);
        v4f f1 = ld4f(xr + k0 + 4);
        v4f f2 = ld4f(xr + k0 + 16);
        v4f f3 = ld4f(xr + k0 + 20);
        v16h a;
#pragma unroll
        for (int i = 0; i < 4; ++i) {
            a[i]      = (_Float16)f0[i];
            a[4 + i]  = (_Float16)f1[i];
            a[8 + i]  = (_Float16)f2[i];
            a[12 + i] = (_Float16)f3[i];
        }
#pragma unroll
        for (int t = 0; t < 4; ++t) {
            const _Float16* wr = W + (size_t)(n0 + t * 16 + l15) * D_ + k0;
            v16h b = cat8(ld8h(wr), ld8h(wr + 16));
            c[t] = wmma16(a, b, c[t]);
        }
    }

    // A 16-row tile never crosses a batch boundary (T_ % 16 == 0): hoist.
    const int head = n0 / 64;
    const int bb   = m0 / T_;
    const int tl0  = (m0 % T_) + hi8;
    if (mat == 2) {
#pragma unroll
        for (int t = 0; t < 4; ++t) {
            const size_t cb = (((size_t)bb * H_ + head) * HD_ + t * 16 + l15) * T_ + tl0;
#pragma unroll
            for (int r = 0; r < 8; ++r)
                vht[cb + r] = (_Float16)c[t][r];
        }
    } else {
        _Float16* dst = (mat == 0) ? qh : kh;
        const size_t rb = (((size_t)bb * H_ + head) * T_ + tl0) * HD_ + l15;
#pragma unroll
        for (int t = 0; t < 4; ++t)
#pragma unroll
            for (int r = 0; r < 8; ++r)
                dst[rb + (size_t)r * HD_ + t * 16] = (_Float16)c[t][r];
    }
}

// ---------------------------------------------------------------------------
// Kernel 3: causal flash attention.
//  Block = one (b,h) x 64 query rows; wave w owns rows q0 = qb*64 + w*16.
//  32-key K/V tiles are staged in LDS by wave 0 via async global->LDS copies
//  (double-buffered, ASYNCcnt-counted), shared by all 4 waves.
//  Per tile & wave: S = Q K^T (4 WMMA), online softmax (shfl over 16-lane
//  halves, v_exp), P re-layout via private LDS tile, O += P V (4 WMMA).
//  Fully-masked trailing tiles of early waves are exact no-ops.
// ---------------------------------------------------------------------------
__global__ void __launch_bounds__(128) attn_kernel(
    const _Float16* __restrict__ qh, const _Float16* __restrict__ kh,
    const _Float16* __restrict__ vht, _Float16* __restrict__ attnbuf)
{
    __shared__ __align__(16) _Float16 kbuf[2][32 * HD_];   // [key][hd]   4KB each
    __shared__ __align__(16) _Float16 vbuf[2][HD_ * 32];   // [hd][key]   4KB each
    __shared__ __align__(16) _Float16 pbuf[4][16 * 32];    // per-wave P  1KB each

    const int lane = threadIdx.x & 31;
    const int warp = threadIdx.x >> 5;
    const int bh = blockIdx.x / (T_ / 64);       // (b,h) of this block
    const int qb = blockIdx.x % (T_ / 64);       // 64-query stripe index
    const int q0 = qb * 64 + warp * 16;

    const int l15  = lane & 15;
    const int koff = (lane & 16) ? 8 : 0;
    const int hi8  = (lane & 16) ? 8 : 0;
    const float scale = 0.125f;                  // 1/sqrt(HD)

    const _Float16* ktile0 = kh  + ((size_t)bh * T_) * HD_;    // + kb*HD_
    const _Float16* vrow0  = vht + ((size_t)bh * HD_) * T_;    // + hd*T_ + kb

    // Q tile: 16 x 64 -> two A operands (K = 0..31, 32..63)
    const _Float16* qr = qh + ((size_t)bh * T_ + q0 + l15) * HD_;
    v16h qa0 = cat8(ld8h(qr + koff),      ld8h(qr + koff + 16));
    v16h qa1 = cat8(ld8h(qr + 32 + koff), ld8h(qr + 32 + koff + 16));

    v8f o0 = {}, o1 = {}, o2 = {}, o3 = {};
    float mrow[8], lrow[8];
#pragma unroll
    for (int r = 0; r < 8; ++r) { mrow[r] = -1e30f; lrow[r] = 0.f; }

    _Float16* pl = pbuf[warp];
    const int jmax = 2 * qb + 1;                 // block-uniform causal bound

    // Prologue: wave 0 issues tiles 0 and 1 (16 async b128 loads per tile).
    if (warp == 0) {
#pragma unroll 1
        for (int j0 = 0; j0 < 2 && j0 <= jmax; ++j0) {
            const int kb = j0 * 32;
#pragma unroll
            for (int c8 = 0; c8 < 8; ++c8) {                 // K tile: 4KB contig
                const int idx = c8 * 32 + lane;
                async_g2l_b128(lds_off(kbuf[j0]) + idx * 16u,
                               (const char*)(ktile0 + (size_t)kb * HD_) + idx * 16);
            }
#pragma unroll
            for (int c8 = 0; c8 < 8; ++c8) {                 // V tile: 64 rows x 64B
                const int idx = c8 * 32 + lane;
                const int hd = idx >> 2, sub = idx & 3;
                async_g2l_b128(lds_off(vbuf[j0]) + idx * 16u,
                               vrow0 + (size_t)hd * T_ + kb + sub * 8);
            }
        }
    }

    for (int j = 0; j <= jmax; ++j) {
        const int p = j & 1;
        const int kb = j * 32;
        // Retire the oldest async batch (16 instrs); keep the prefetch in flight.
        if (warp == 0) {
            if (j < jmax) asm volatile("s_wait_asynccnt 16" ::: "memory");
            else          asm volatile("s_wait_asynccnt 0"  ::: "memory");
        }
        __syncthreads();

        // --- scores: S[16x32] = Q[16x64] * K^T from LDS K tile ---
        v8f s0 = {}, s1 = {};
        {
            const _Float16* k0r = kbuf[p] + (size_t)(l15)      * HD_;
            const _Float16* k1r = kbuf[p] + (size_t)(16 + l15) * HD_;
            v16h b;
            b = cat8(ld8h(k0r + koff),      ld8h(k0r + koff + 16));      s0 = wmma16(qa0, b, s0);
            b = cat8(ld8h(k0r + 32 + koff), ld8h(k0r + 32 + koff + 16)); s0 = wmma16(qa1, b, s0);
            b = cat8(ld8h(k1r + koff),      ld8h(k1r + koff + 16));      s1 = wmma16(qa0, b, s1);
            b = cat8(ld8h(k1r + 32 + koff), ld8h(k1r + 32 + koff + 16)); s1 = wmma16(qa1, b, s1);
        }
        // --- online softmax over the 32 keys of this tile ---
        const int key0 = kb + l15, key1 = kb + 16 + l15;
#pragma unroll
        for (int r = 0; r < 8; ++r) {
            const int qg = q0 + r + hi8;
            float a0 = s0[r] * scale; if (key0 > qg) a0 = -1e30f;
            float a1 = s1[r] * scale; if (key1 > qg) a1 = -1e30f;
            float rmax = fmaxf(a0, a1);
#pragma unroll
            for (int msk = 1; msk < 16; msk <<= 1)
                rmax = fmaxf(rmax, __shfl_xor(rmax, msk, 32));
            const float mnew = fmaxf(mrow[r], rmax);
            const float p0 = __expf(a0 - mnew);
            const float p1 = __expf(a1 - mnew);
            float rsum = p0 + p1;
#pragma unroll
            for (int msk = 1; msk < 16; msk <<= 1)
                rsum += __shfl_xor(rsum, msk, 32);
            const float alpha = __expf(mrow[r] - mnew);
            lrow[r] = lrow[r] * alpha + rsum;
            mrow[r] = mnew;
            o0[r] *= alpha; o1[r] *= alpha; o2[r] *= alpha; o3[r] *= alpha;
            pl[(r + hi8) * 32 + l15]      = (_Float16)p0;
            pl[(r + hi8) * 32 + l15 + 16] = (_Float16)p1;
        }
        // wave-internal cross-lane RAW through LDS: wait own DS stores
        asm volatile("s_wait_dscnt 0" ::: "memory");
        v16h pa = cat8(ld8h(pl + l15 * 32 + koff),
                       ld8h(pl + l15 * 32 + koff + 16));
        // --- O += P[16x32] * V[32x64] from LDS V tile ([hd][key]) ---
        {
            const _Float16* v0 = vbuf[p] + (size_t)( 0 + l15) * 32 + koff;
            const _Float16* v1 = vbuf[p] + (size_t)(16 + l15) * 32 + koff;
            const _Float16* v2 = vbuf[p] + (size_t)(32 + l15) * 32 + koff;
            const _Float16* v3 = vbuf[p] + (size_t)(48 + l15) * 32 + koff;
            o0 = wmma16(pa, cat8(ld8h(v0), ld8h(v0 + 16)), o0);
            o1 = wmma16(pa, cat8(ld8h(v1), ld8h(v1 + 16)), o1);
            o2 = wmma16(pa, cat8(ld8h(v2), ld8h(v2 + 16)), o2);
            o3 = wmma16(pa, cat8(ld8h(v3), ld8h(v3 + 16)), o3);
        }
        __syncthreads();   // all waves done reading buffer p
        // Prefetch tile j+2 into the buffer just released.
        if (warp == 0 && j + 2 <= jmax) {
            const int kb2 = (j + 2) * 32;
#pragma unroll
            for (int c8 = 0; c8 < 8; ++c8) {
                const int idx = c8 * 32 + lane;
                async_g2l_b128(lds_off(kbuf[p]) + idx * 16u,
                               (const char*)(ktile0 + (size_t)kb2 * HD_) + idx * 16);
            }
#pragma unroll
            for (int c8 = 0; c8 < 8; ++c8) {
                const int idx = c8 * 32 + lane;
                const int hd = idx >> 2, sub = idx & 3;
                async_g2l_b128(lds_off(vbuf[p]) + idx * 16u,
                               vrow0 + (size_t)hd * T_ + kb2 + sub * 8);
            }
        }
    }

    // --- finalize: divide by row sums, store f16 [b,t,d] for output proj ---
    const int b_ = bh / H_, h_ = bh % H_;
#pragma unroll
    for (int r = 0; r < 8; ++r) {
        const float inv = 1.0f / lrow[r];
        const int t = q0 + r + hi8;
        const size_t base = ((size_t)b_ * T_ + t) * D_ + (size_t)h_ * HD_;
        attnbuf[base +  0 + l15] = (_Float16)(o0[r] * inv);
        attnbuf[base + 16 + l15] = (_Float16)(o1[r] * inv);
        attnbuf[base + 32 + l15] = (_Float16)(o2[r] * inv);
        attnbuf[base + 48 + l15] = (_Float16)(o3[r] * inv);
    }
}

// ---------------------------------------------------------------------------
// Kernel 4: output projection  out = attn @ Wo^T  (f16 x f16 -> f32)
// ---------------------------------------------------------------------------
__global__ void __launch_bounds__(128) outproj_kernel(
    const _Float16* __restrict__ A, const _Float16* __restrict__ Wo16,
    float* __restrict__ Y)
{
    const int lane = threadIdx.x & 31;
    const int warp = threadIdx.x >> 5;
    const int wave = blockIdx.x * 4 + warp;
    const int MT = (B_ * T_) / 16;
    const int m0 = (wave % MT) * 16;
    const int n0 = (wave / MT) * 64;

    const int l15  = lane & 15;
    const int koff = (lane & 16) ? 8 : 0;
    const int hi8  = (lane & 16) ? 8 : 0;

    v8f c[4] = {v8f{}, v8f{}, v8f{}, v8f{}};
    const _Float16* ar = A + (size_t)(m0 + l15) * D_;
#pragma unroll 1
    for (int ks = 0; ks < D_ / 32; ++ks) {
        const int k0 = ks * 32 + koff;
        v16h a = cat8(ld8h(ar + k0), ld8h(ar + k0 + 16));
#pragma unroll
        for (int t = 0; t < 4; ++t) {
            const _Float16* wr = Wo16 + (size_t)(n0 + t * 16 + l15) * D_ + k0;
            v16h b = cat8(ld8h(wr), ld8h(wr + 16));
            c[t] = wmma16(a, b, c[t]);
        }
    }
#pragma unroll
    for (int t = 0; t < 4; ++t)
#pragma unroll
        for (int r = 0; r < 8; ++r)
            Y[(size_t)(m0 + r + hi8) * D_ + n0 + t * 16 + l15] = c[t][r];
}

// ---------------------------------------------------------------------------
extern "C" void kernel_launch(void* const* d_in, const int* in_sizes, int n_in,
                              void* d_out, int out_size, void* d_ws, size_t ws_size,
                              hipStream_t stream) {
    const float* q  = (const float*)d_in[0];
    const float* k  = (const float*)d_in[1];
    const float* v  = (const float*)d_in[2];
    // d_in[3] is the causal mask; causality is applied analytically in attn_kernel.
    const float* Wq = (const float*)d_in[4];
    const float* Wk = (const float*)d_in[5];
    const float* Wv = (const float*)d_in[6];
    const float* Wo = (const float*)d_in[7];
    float* out = (float*)d_out;

    char* ws = (char*)d_ws;
    const size_t szW16  = (size_t)D_ * D_ * sizeof(_Float16);            // 2 MB each
    const size_t szHead = (size_t)B_ * H_ * T_ * HD_ * sizeof(_Float16); // 16 MB each
    _Float16* W16 = (_Float16*)ws;                        // 4 matrices (q,k,v,o)
    _Float16* qh  = (_Float16*)(ws + 4 * szW16);
    _Float16* kh  = (_Float16*)(ws + 4 * szW16 + szHead);
    _Float16* vht = (_Float16*)(ws + 4 * szW16 + 2 * szHead);
    _Float16* ab  = (_Float16*)(ws + 4 * szW16 + 3 * szHead);

    cvt_w_kernel<<<dim3((4u << 20) / 256), 256, 0, stream>>>(Wq, Wk, Wv, Wo, W16);
    proj_kernel<<<dim3(2048, 3), 128, 0, stream>>>(q, k, v, W16, qh, kh, vht);
    attn_kernel<<<dim3(2048), 128, 0, stream>>>(qh, kh, vht, ab);
    outproj_kernel<<<dim3(2048), 128, 0, stream>>>(ab, W16 + 3 * (size_t)D_ * D_, out);
}
